// SparseFeatureAttention_86028194939125
// MI455X (gfx1250) — matrix operand order repaired and checked by reading, not compiled, not run on hardware
//
#include <hip/hip_runtime.h>
#include <hip/hip_bf16.h>
#include <stdint.h>

typedef __bf16 bf16;
typedef __attribute__((ext_vector_type(16))) __bf16 v16bf;
typedef __attribute__((ext_vector_type(8)))  __bf16 v8bf;
typedef __attribute__((ext_vector_type(8)))  float   v8f;

// ---------------------------------------------------------------------------
// WMMA fragment helpers (CDNA5 wave32 layouts, 16x16x32 bf16)
// A-matrix 16x32: lane&15 = row M; lane>>4 selects K-half; per lane two
//   contiguous 8-element K runs at k0 and k0+16  (k0 = k + (lane>>4)*8)
// B-matrix 32x16 (column-major source): lane&15 = col N; per lane two
//   contiguous 8-element K runs at k0 and k0+16  (k0 = k + (lane>>4)*8)
// C/D 16x16 f32: element e -> row = e + 8*(lane>>4), col = lane&15
// ---------------------------------------------------------------------------
__device__ __forceinline__ v16bf wmma_load_a(const bf16* __restrict__ base,
                                             int ld, int row0, int k, int lane) {
  const bf16* p = base + (size_t)(row0 + (lane & 15)) * ld + k + ((lane >> 4) << 3);
  v8bf lo = *(const v8bf*)p;          // K = k0 .. k0+7
  v8bf hi = *(const v8bf*)(p + 16);   // K = k0+16 .. k0+23
  return __builtin_shufflevector(lo, hi, 0,1,2,3,4,5,6,7,8,9,10,11,12,13,14,15);
}

__device__ __forceinline__ v16bf wmma_load_b(const bf16* __restrict__ base,
                                             int ld, int col0, int k, int lane) {
  const bf16* p = base + (size_t)(col0 + (lane & 15)) * ld + k + ((lane >> 4) << 3);
  v8bf lo = *(const v8bf*)p;
  v8bf hi = *(const v8bf*)(p + 16);
  return __builtin_shufflevector(lo, hi, 0,1,2,3,4,5,6,7,8,9,10,11,12,13,14,15);
}

__device__ __forceinline__ v8f wmma_bf16(v16bf a, v16bf b, v8f c) {
  return __builtin_amdgcn_wmma_f32_16x16x32_bf16(false, a, false, b,
                                                 (short)0, c, false, false);
}

// ---------------------------------------------------------------------------
// Register-resident top-512-of-1024 threshold + softmax for one row.
// v[] holds the row striped: lane owns elements {lane + 32*i}. After return,
// v[] holds normalized probabilities (zeros below threshold).
// ---------------------------------------------------------------------------
__device__ __forceinline__ void topk_softmax_row(float (&v)[32]) {
  float mx = v[0], mn = v[0];
#pragma unroll
  for (int i = 1; i < 32; ++i) { mx = fmaxf(mx, v[i]); mn = fminf(mn, v[i]); }
#pragma unroll
  for (int o = 16; o; o >>= 1) {
    mx = fmaxf(mx, __shfl_xor(mx, o, 32));
    mn = fminf(mn, __shfl_xor(mn, o, 32));
  }

  // bisection: largest t with count(v >= t) >= 512  (pure VALU, no LDS)
  float lo = mn, hi = mx;
  for (int it = 0; it < 24; ++it) {
    float mid = 0.5f * (lo + hi);
    int cnt = 0;
#pragma unroll
    for (int i = 0; i < 32; ++i) cnt += (v[i] >= mid) ? 1 : 0;
#pragma unroll
    for (int o = 16; o; o >>= 1) cnt += __shfl_xor(cnt, o, 32);
    if (cnt >= 512) lo = mid; else hi = mid;
  }
  const float thr = lo;

  float denom = 0.f;
#pragma unroll
  for (int i = 0; i < 32; ++i) {
    float e = (v[i] >= thr) ? __expf(v[i] - mx) : 0.f;
    v[i] = e;
    denom += e;
  }
#pragma unroll
  for (int o = 16; o; o >>= 1) denom += __shfl_xor(denom, o, 32);
  float inv = 1.f / denom;
#pragma unroll
  for (int i = 0; i < 32; ++i) v[i] *= inv;
}

// ---------------------------------------------------------------------------
// Precision conversion / layout kernels
// ---------------------------------------------------------------------------
__global__ void cvt_f32_to_bf16(const float* __restrict__ s, bf16* __restrict__ d, size_t n) {
  size_t i = (size_t)blockIdx.x * blockDim.x + threadIdx.x;
  size_t stride = (size_t)gridDim.x * blockDim.x;
  for (; i < n; i += stride) d[i] = (__bf16)s[i];
}

// W[k][n] fp32 (D x D) -> Wt[n][k] bf16 (column-major for WMMA B fragments)
__global__ void cvt_transpose_bf16(const float* __restrict__ W, bf16* __restrict__ Wt, int D) {
  int idx = blockIdx.x * blockDim.x + threadIdx.x;
  if (idx < D * D) {
    int k = idx / D, n = idx % D;
    Wt[(size_t)n * D + k] = (__bf16)W[idx];
  }
}

// ---------------------------------------------------------------------------
// Generic bf16 WMMA GEMM:  C[M x N] = A[M x K] * Bt^T + bias
//   A row-major bf16, Bt column-major bf16 (i.e. Bt[n][k])
//   block = 256 thr (8 waves, 2x4), tile 32 x 512; each wave 16 x 128 (8 accs)
//   mode 0: fp32 row-major out    (final projection -> d_out)
//   mode 1: bf16 row-major out    (Q, K)
//   mode 2: bf16 head-transposed  (V -> Vt[b][h][d][n], contiguous in n)
// ---------------------------------------------------------------------------
__global__ __launch_bounds__(256)
void gemm_wmma_bf16(const bf16* __restrict__ A, const bf16* __restrict__ Bt,
                    const float* __restrict__ bias,
                    float* __restrict__ outF, bf16* __restrict__ outB,
                    int M, int N, int K, int mode, int batchN, int H, int hd) {
  const int lane = threadIdx.x & 31;
  const int wave = threadIdx.x >> 5;
  const int wr = wave >> 2, wc = wave & 3;
  const int row0 = blockIdx.x * 32 + wr * 16;
  const int col0 = blockIdx.y * 512 + wc * 128;

  v8f acc[8] = {v8f{}, v8f{}, v8f{}, v8f{}, v8f{}, v8f{}, v8f{}, v8f{}};
  for (int k = 0; k < K; k += 32) {
    v16bf a = wmma_load_a(A, K, row0, k, lane);
#pragma unroll
    for (int j = 0; j < 8; ++j) {
      v16bf b = wmma_load_b(Bt, K, col0 + 16 * j, k, lane);
      acc[j] = wmma_bf16(a, b, acc[j]);
    }
  }

#pragma unroll
  for (int j = 0; j < 8; ++j) {
    int col = col0 + 16 * j + (lane & 15);
    float bv = bias[col];
#pragma unroll
    for (int e = 0; e < 8; ++e) {
      int row = row0 + ((lane >> 4) << 3) + e;
      float v = acc[j][e] + bv;
      if (mode == 0) {
        outF[(size_t)row * N + col] = v;
      } else if (mode == 1) {
        outB[(size_t)row * N + col] = (__bf16)v;
      } else {
        int bb = row / batchN, nn = row % batchN;
        int hh = col / hd,     dd = col % hd;
        outB[((size_t)(bb * H + hh) * hd + dd) * batchN + nn] = (__bf16)v;
      }
    }
  }
}

// ---------------------------------------------------------------------------
// Fused sparse attention: one block = 16 queries of one (b, h).
//   Phase 1: scores 16x1024 via WMMA -> LDS (fp32, 64 KB)
//   Phase 2: rows live in registers: top-512 threshold (bisection), softmax;
//            probs written back to LDS as bf16 (aliases low 32 KB of buffer)
//   Phase 3: probs(16x1024 bf16, LDS) x V(1024x128) via WMMA, bf16 out
// ---------------------------------------------------------------------------
__global__ __launch_bounds__(256)
void attn_wmma(const bf16* __restrict__ Q, const bf16* __restrict__ Kmat,
               const bf16* __restrict__ Vt, bf16* __restrict__ O,
               int Nn, int D, int H, float scale) {
  const int hd = D / H;                       // 128
  __shared__ float s_sc[16 * 1024];           // 64 KB scores
  bf16* s_pb = (bf16*)s_sc;                   // bf16 probs alias, low 32 KB

  const int b = blockIdx.z, h = blockIdx.y, qt = blockIdx.x;
  const int lane = threadIdx.x & 31;
  const int wave = threadIdx.x >> 5;          // 8 waves

  const bf16* Qb = Q    + ((size_t)b * Nn + qt * 16) * D + h * hd;
  const bf16* Kb = Kmat + (size_t)b * Nn * D + h * hd;

  // ---- Phase 1: scores = (Q16x128 . K^T) * scale, wave w -> keys [w*128, +128)
  v16bf aQ[4];
#pragma unroll
  for (int t = 0; t < 4; ++t) aQ[t] = wmma_load_a(Qb, D, 0, t * 32, lane);

  const int kb = wave * 128;
  for (int ct = 0; ct < 8; ++ct) {
    v8f c = v8f{};
#pragma unroll
    for (int t = 0; t < 4; ++t) {
      v16bf bk = wmma_load_b(Kb, D, kb + ct * 16, t * 32, lane);
      c = wmma_bf16(aQ[t], bk, c);
    }
    int col = kb + ct * 16 + (lane & 15);
#pragma unroll
    for (int e = 0; e < 8; ++e) {
      int r = ((lane >> 4) << 3) + e;
      s_sc[r * 1024 + col] = c[e] * scale;
    }
  }
  __syncthreads();

  // ---- Phase 2: wave w owns rows 2w, 2w+1; rows -> registers (32 VGPR each)
  const int r0 = wave * 2, r1 = wave * 2 + 1;
  float v0[32], v1[32];
#pragma unroll
  for (int i = 0; i < 32; ++i) v0[i] = s_sc[r0 * 1024 + lane + 32 * i];
#pragma unroll
  for (int i = 0; i < 32; ++i) v1[i] = s_sc[r1 * 1024 + lane + 32 * i];
  __syncthreads();            // all fp32 reads done before bf16 alias writes

  topk_softmax_row(v0);       // pure VALU + shuffles, zero LDS traffic
  topk_softmax_row(v1);

#pragma unroll
  for (int i = 0; i < 32; ++i) s_pb[r0 * 1024 + lane + 32 * i] = (__bf16)v0[i];
#pragma unroll
  for (int i = 0; i < 32; ++i) s_pb[r1 * 1024 + lane + 32 * i] = (__bf16)v1[i];
  __syncthreads();

  // ---- Phase 3: out(16 x 128) = probs(16x1024) x V ; wave w -> d in [w*16,+16)
  const bf16* Vb = Vt + (size_t)(b * H + h) * hd * Nn;   // [d][n], n contiguous
  v8f c = v8f{};
  for (int kk = 0; kk < 1024; kk += 32) {
    v16bf a  = wmma_load_a(s_pb, 1024, 0, kk, lane);     // 2x ds_load_b128
    v16bf bv = wmma_load_b(Vb, Nn, wave * 16, kk, lane);
    c = wmma_bf16(a, bv, c);
  }
  bf16* Ob = O + ((size_t)b * Nn + qt * 16) * D + h * hd;
  int col = wave * 16 + (lane & 15);
#pragma unroll
  for (int e = 0; e < 8; ++e) {
    int r = ((lane >> 4) << 3) + e;
    Ob[(size_t)r * D + col] = (__bf16)c[e];
  }
}

// ---------------------------------------------------------------------------
extern "C" void kernel_launch(void* const* d_in, const int* in_sizes, int n_in,
                              void* d_out, int out_size, void* d_ws, size_t ws_size,
                              hipStream_t stream) {
  (void)in_sizes; (void)n_in; (void)out_size; (void)ws_size;

  constexpr int B = 16, N = 1024, D = 1024, H = 8, hd = D / H;
  const size_t MN = (size_t)B * N;            // 16384 rows

  const float* x  = (const float*)d_in[0];
  const float* Wq = (const float*)d_in[1];
  const float* bq = (const float*)d_in[2];
  const float* Wk = (const float*)d_in[3];
  const float* bk = (const float*)d_in[4];
  const float* Wv = (const float*)d_in[5];
  const float* bv = (const float*)d_in[6];
  const float* Wo = (const float*)d_in[7];
  const float* bo = (const float*)d_in[8];
  float* out = (float*)d_out;

  // workspace carve-out (256-byte aligned)
  uint8_t* cur = (uint8_t*)d_ws;
  auto carve = [&](size_t bytes) -> void* {
    void* r = cur;
    cur += (bytes + 255) & ~(size_t)255;
    return r;
  };
  bf16* xb  = (bf16*)carve(MN * D * sizeof(bf16));
  bf16* wqt = (bf16*)carve((size_t)D * D * sizeof(bf16));
  bf16* wkt = (bf16*)carve((size_t)D * D * sizeof(bf16));
  bf16* wvt = (bf16*)carve((size_t)D * D * sizeof(bf16));
  bf16* wot = (bf16*)carve((size_t)D * D * sizeof(bf16));
  bf16* Qb  = (bf16*)carve(MN * D * sizeof(bf16));
  bf16* Kb  = (bf16*)carve(MN * D * sizeof(bf16));
  bf16* Vtb = (bf16*)carve(MN * D * sizeof(bf16));
  bf16* Ab  = (bf16*)carve(MN * D * sizeof(bf16));

  // 1) precision / layout conversion
  cvt_f32_to_bf16<<<2048, 256, 0, stream>>>(x, xb, MN * D);
  const int tgrid = (D * D + 255) / 256;
  cvt_transpose_bf16<<<tgrid, 256, 0, stream>>>(Wq, wqt, D);
  cvt_transpose_bf16<<<tgrid, 256, 0, stream>>>(Wk, wkt, D);
  cvt_transpose_bf16<<<tgrid, 256, 0, stream>>>(Wv, wvt, D);
  cvt_transpose_bf16<<<tgrid, 256, 0, stream>>>(Wo, wot, D);

  // 2) Q/K/V projections (bf16 WMMA, fp32 accumulate)
  dim3 ggrid(MN / 32, D / 512);
  gemm_wmma_bf16<<<ggrid, 256, 0, stream>>>(xb, wqt, bq, nullptr, Qb,
                                            (int)MN, D, D, 1, N, H, hd);
  gemm_wmma_bf16<<<ggrid, 256, 0, stream>>>(xb, wkt, bk, nullptr, Kb,
                                            (int)MN, D, D, 1, N, H, hd);
  gemm_wmma_bf16<<<ggrid, 256, 0, stream>>>(xb, wvt, bv, nullptr, Vtb,
                                            (int)MN, D, D, 2, N, H, hd);

  // 3) fused sparse attention (scores never touch HBM)
  const float scale = 1.0f / sqrtf((float)hd);
  attn_wmma<<<dim3(N / 16, H, B), 256, 0, stream>>>(Qb, Kb, Vtb, Ab,
                                                    N, D, H, scale);

  // 4) output projection -> fp32 d_out
  gemm_wmma_bf16<<<ggrid, 256, 0, stream>>>(Ab, wot, bo, out, nullptr,
                                            (int)MN, D, D, 0, N, H, hd);
}
